// SpGraphAttentionLayer_21474836480626
// MI455X (gfx1250) — matrix operand order repaired and checked by reading, not compiled, not run on hardware
//
#include <hip/hip_runtime.h>
#include <math.h>

typedef float v2f  __attribute__((ext_vector_type(2)));
typedef float v8f  __attribute__((ext_vector_type(8)));
typedef _Float16 v16h __attribute__((ext_vector_type(16)));

#define N_NODES 100000
#define N_EDGES 1600000
#define IN_F    128
#define OUT_F   64
#define EMB     64
#define ATT     32
#define MAX_R   5
#define ALPHA   0.2f

// ---------------------------------------------------------------------------
// Kernel: zero a float span (hprime + rowsum, laid out contiguously)
// ---------------------------------------------------------------------------
__global__ void zero_f32(float* __restrict__ p, int n) {
    int i = blockIdx.x * blockDim.x + threadIdx.x;
    if (i < n) p[i] = 0.0f;
}

// ---------------------------------------------------------------------------
// Kernel: fold attention vectors through the tiny projections:
//   re_v[k] = sum_t re_W[k][t] * a[2*OUT_F + t]          (k < EMB)
//   ra_v[r] = sum_t ra_W[r][t] * a[2*OUT_F + ATT + t]    (r < MAX_R)
// ---------------------------------------------------------------------------
__global__ void prep_vecs(const float* __restrict__ a,
                          const float* __restrict__ re_W,
                          const float* __restrict__ ra_W,
                          float* __restrict__ re_v,
                          float* __restrict__ ra_v) {
    int t = threadIdx.x;  // 64 threads
    if (t < EMB) {
        float s = 0.0f;
        #pragma unroll
        for (int j = 0; j < ATT; ++j) s += re_W[t * ATT + j] * a[2 * OUT_F + j];
        re_v[t] = s;
    }
    if (t < MAX_R) {
        float s = 0.0f;
        #pragma unroll
        for (int j = 0; j < ATT; ++j) s += ra_W[t * ATT + j] * a[2 * OUT_F + ATT + j];
        ra_v[t] = s;
    }
}

// ---------------------------------------------------------------------------
// Kernel: h = inputs(100000x128) @ W(128x64) via CDNA5 WMMA.
// One wave per 16-row tile; computes all 4 16-col tiles (A-frag reuse).
// f32 WMMA 16x16x4 layouts (ISA 7.12.2, 32-bit operands, wave32):
//   A 16x4 : lane L (m=L&15): v0 = A[m][k + 2*(L>>4)], v1 = A[m][k + 2*(L>>4)+1]
//   B 4x16 : lane L (n=L&15): v0 = B[k + 2*(L>>4)][n], v1 = B[k + 2*(L>>4)+1][n]
//   C 16x16: VGPR j: M = j + 8*(L>>4), N = L&15
// ---------------------------------------------------------------------------
__global__ __launch_bounds__(256) void gemm_h_wmma(const float* __restrict__ A,
                                                   const float* __restrict__ W,
                                                   float* __restrict__ H) {
    int wave = (int)((blockIdx.x * blockDim.x + threadIdx.x) >> 5);
    int lane = threadIdx.x & 31;
    const int nRowTiles = N_NODES / 16;  // 6250 exactly
    if (wave >= nRowTiles) return;

    int row0 = wave * 16;
    int m    = lane & 15;
    int n    = lane & 15;
    int kg   = (lane >> 4) * 2;  // 0 or 2 : which K-pair this half-wave owns

    v8f acc[4] = {v8f{}, v8f{}, v8f{}, v8f{}};

#if __has_builtin(__builtin_amdgcn_wmma_f32_16x16x4_f32)
    const float* arow = A + (size_t)(row0 + m) * IN_F;
    for (int k = 0; k < IN_F; k += 4) {
        v2f af = *(const v2f*)(arow + k + kg);  // 8B aligned: k+kg even
        const float* b0 = W + (size_t)(k + kg) * OUT_F;
        const float* b1 = b0 + OUT_F;
        #pragma unroll
        for (int c = 0; c < 4; ++c) {
            v2f bf;
            bf.x = b0[n + 16 * c];
            bf.y = b1[n + 16 * c];
            acc[c] = __builtin_amdgcn_wmma_f32_16x16x4_f32(
                false, af, false, bf, (short)0, acc[c], false, false);
        }
    }
#else
    // Fallback: f16 WMMA 16x16x32 (codegen-confirmed builtin), f32 accumulate.
    int hi = lane >> 4;
    const float* arow = A + (size_t)(row0 + m) * IN_F;
    for (int k = 0; k < IN_F; k += 32) {
        v16h af;
        const float* ap = arow + k + 8 * hi;
        #pragma unroll
        for (int i = 0; i < 8; ++i) af[i] = (_Float16)ap[i];
        #pragma unroll
        for (int i = 0; i < 8; ++i) af[8 + i] = (_Float16)ap[16 + i];
        #pragma unroll
        for (int c = 0; c < 4; ++c) {
            v16h bf;
            #pragma unroll
            for (int i = 0; i < 16; ++i)
                bf[i] = (_Float16)W[(size_t)(k + 8 * hi + i + (i & 8)) * OUT_F + n + 16 * c];
            acc[c] = __builtin_amdgcn_wmma_f32_16x16x32_f16(
                false, af, false, bf, (short)0, acc[c], false, false);
        }
    }
#endif

    int mrow = 8 * (lane >> 4);
    #pragma unroll
    for (int j = 0; j < 8; ++j) {
        float* o = H + (size_t)(row0 + mrow + j) * OUT_F + n;
        o[0]  = acc[0][j];
        o[16] = acc[1][j];
        o[32] = acc[2][j];
        o[48] = acc[3][j];
    }
}

// ---------------------------------------------------------------------------
// Kernel: per-node attention scalars  hs[i] = h[i]·a_src,  hd[i] = h[i]·a_dst
// ---------------------------------------------------------------------------
__global__ void node_scores(const float* __restrict__ h,
                            const float* __restrict__ a,
                            float* __restrict__ hs,
                            float* __restrict__ hd) {
    int i = blockIdx.x * blockDim.x + threadIdx.x;
    if (i >= N_NODES) return;
    const float* hr = h + (size_t)i * OUT_F;
    float s0 = 0.0f, s1 = 0.0f;
    #pragma unroll
    for (int j = 0; j < OUT_F; ++j) {
        float v = hr[j];
        s0 += v * a[j];
        s1 += v * a[OUT_F + j];
    }
    hs[i] = s0;
    hd[i] = s1;
}

// ---------------------------------------------------------------------------
// Kernel: one wave32 per edge.
//   score = hs[src] + hd[dst] + review[e]·re_v + rating[e]·ra_v
//   w     = exp(leaky_relu(score))
//   rowsum[src] += w ;  hprime[src][:] += w * h[dst][:]
// review read is wave-coalesced (lanes cover the 64-float row in 2 loads);
// h gathers and hprime atomics hit the L2-resident 25.6MB arrays.
// ---------------------------------------------------------------------------
__global__ __launch_bounds__(256) void edge_scatter(const int* __restrict__ src,
                                                    const int* __restrict__ dst,
                                                    const float* __restrict__ review,
                                                    const float* __restrict__ rating,
                                                    const float* __restrict__ h,
                                                    const float* __restrict__ hs,
                                                    const float* __restrict__ hd,
                                                    const float* __restrict__ re_v,
                                                    const float* __restrict__ ra_v,
                                                    float* __restrict__ hprime,
                                                    float* __restrict__ rowsum) {
    int e    = (int)((blockIdx.x * blockDim.x + threadIdx.x) >> 5);
    int lane = threadIdx.x & 31;
    if (e >= N_EDGES) return;

    int s = src[e];
    int d = dst[e];

    const float* rv = review + (size_t)e * EMB;
    float p = rv[lane] * re_v[lane] + rv[lane + 32] * re_v[lane + 32];
    if (lane < MAX_R) p += rating[(size_t)e * MAX_R + lane] * ra_v[lane];

    // wave32 butterfly reduction
    #pragma unroll
    for (int off = 16; off >= 1; off >>= 1) p += __shfl_xor(p, off, 32);

    float score = p + hs[s] + hd[d];
    float lr    = score > 0.0f ? score : ALPHA * score;
    float w     = __expf(lr);

    if (lane == 0) atomicAdd(rowsum + s, w);

    const float* hdr = h + (size_t)d * OUT_F;
    float*       hp  = hprime + (size_t)s * OUT_F;
    atomicAdd(hp + lane,      w * hdr[lane]);
    atomicAdd(hp + lane + 32, w * hdr[lane + 32]);
}

// ---------------------------------------------------------------------------
// Kernel: out = elu( hprime / (rowsum + 1e-10) + h )
// ---------------------------------------------------------------------------
__global__ void finalize(const float* __restrict__ h,
                         const float* __restrict__ hprime,
                         const float* __restrict__ rowsum,
                         float* __restrict__ out) {
    int i = blockIdx.x * blockDim.x + threadIdx.x;
    if (i >= N_NODES * OUT_F) return;
    int node = i >> 6;  // OUT_F == 64
    float v = hprime[i] / (rowsum[node] + 1e-10f) + h[i];
    out[i] = v > 0.0f ? v : expm1f(v);
}

// ---------------------------------------------------------------------------
// Launch
// ---------------------------------------------------------------------------
extern "C" void kernel_launch(void* const* d_in, const int* in_sizes, int n_in,
                              void* d_out, int out_size, void* d_ws, size_t ws_size,
                              hipStream_t stream) {
    const float* inputs = (const float*)d_in[0];
    const int*   edge   = (const int*)  d_in[1];
    const float* review = (const float*)d_in[2];
    const float* rating = (const float*)d_in[3];
    const float* W      = (const float*)d_in[4];
    const float* a      = (const float*)d_in[5];
    const float* re_W   = (const float*)d_in[6];
    const float* ra_W   = (const float*)d_in[7];
    float*       out    = (float*)d_out;

    const int* src = edge;            // edge[0]
    const int* dst = edge + N_EDGES;  // edge[1]

    // Workspace layout (floats). hprime and rowsum are adjacent for one zero pass.
    float* ws     = (float*)d_ws;
    float* h      = ws;               ws += (size_t)N_NODES * OUT_F;  // 6.4M
    float* hprime = ws;               ws += (size_t)N_NODES * OUT_F;  // 6.4M
    float* rowsum = ws;               ws += N_NODES;                  // 100K
    float* hs     = ws;               ws += N_NODES;
    float* hd     = ws;               ws += N_NODES;
    float* re_v   = ws;               ws += EMB;
    float* ra_v   = ws;               ws += 8;

    const int zeroN = N_NODES * OUT_F + N_NODES;  // hprime + rowsum
    zero_f32<<<(zeroN + 255) / 256, 256, 0, stream>>>(hprime, zeroN);

    prep_vecs<<<1, 64, 0, stream>>>(a, re_W, ra_W, re_v, ra_v);

    const int nRowTiles = N_NODES / 16;  // 6250
    gemm_h_wmma<<<(nRowTiles + 7) / 8, 256, 0, stream>>>(inputs, W, h);

    node_scores<<<(N_NODES + 255) / 256, 256, 0, stream>>>(h, a, hs, hd);

    // 8 waves per 256-thread block, one wave per edge
    edge_scatter<<<(N_EDGES + 7) / 8, 256, 0, stream>>>(
        src, dst, review, rating, h, hs, hd, re_v, ra_v, hprime, rowsum);

    finalize<<<(N_NODES * OUT_F + 255) / 256, 256, 0, stream>>>(h, hprime, rowsum, out);
}
